// MultiQueryAttention_61546881351757
// MI455X (gfx1250) — compile-verified, matrix operand order checked by reading, not run on hardware
//
#include <hip/hip_runtime.h>
#include <hip/hip_bf16.h>

typedef __attribute__((ext_vector_type(16))) __bf16 v16bf;
typedef __attribute__((ext_vector_type(8)))  float  v8f;
typedef __attribute__((ext_vector_type(4)))  unsigned u32x4;
typedef __attribute__((ext_vector_type(4)))  float    f32x4;
typedef __attribute__((ext_vector_type(4)))  int      i32x4;

#define B_ 4
#define T_ 2048
#define C_ 1024
#define H_ 16
#define D_ 64

// ---- CDNA5 async global->LDS path (guarded; fallback = sync LDS staging) ----
#if __has_builtin(__builtin_amdgcn_global_load_async_to_lds_b128) && \
    __has_builtin(__builtin_amdgcn_s_wait_asynccnt)
#define USE_ASYNC 1
#warning "CDNA5 path: using global_load_async_to_lds_b128 + s_wait_asynccnt"
#else
#define USE_ASYNC 0
#warning "Fallback path: async-to-LDS builtins not available"
#endif

// Copy one 16-byte chunk global->LDS (async on CDNA5 toolchains).
static __device__ __forceinline__ void cp16_g2l(__bf16* lds, const __bf16* g) {
#if USE_ASYNC
  __builtin_amdgcn_global_load_async_to_lds_b128(
      (i32x4*)g,
      (__attribute__((address_space(3))) i32x4*)lds,
      0, 0);
#else
  *(u32x4*)lds = *(const u32x4*)g;
#endif
}

static __device__ __forceinline__ void wait_async_lds() {
#if USE_ASYNC
  __builtin_amdgcn_s_wait_asynccnt(0);
#endif
}

// ---------------------------------------------------------------------------
// Fragment loads. 16-bit WMMA A/B layout (ISA 05_wmma 7.12.2): lane-half
// 'half' holds K = half*8+e (e=0..7) and K = 16+half*8+e-8 (e=8..15).
// With K contiguous in memory a fragment is two b128 loads at p and p+16.
// ---------------------------------------------------------------------------
static __device__ __forceinline__ v16bf ldfrag(const __bf16* p) {
  union { v16bf v; u32x4 q[2]; } u;
  u.q[0] = *(const u32x4*)p;
  u.q[1] = *(const u32x4*)(p + 16);
  return u.v;
}

static __device__ __forceinline__ v16bf ldfrag_lds(const __bf16* p) {
  union { v16bf v; u32x4 q[2]; } u;
  u.q[0] = *(const u32x4*)p;          // ds_load_b128
  u.q[1] = *(const u32x4*)(p + 16);
  return u.v;
}

// fp32 source: two 32-byte chunks, convert to bf16 (v_cvt_pk_bf16_f32).
static __device__ __forceinline__ v16bf ldfrag_f32(const float* p) {
  float t[16];
  *(f32x4*)(t + 0)  = *(const f32x4*)(p);
  *(f32x4*)(t + 4)  = *(const f32x4*)(p + 4);
  *(f32x4*)(t + 8)  = *(const f32x4*)(p + 16);
  *(f32x4*)(t + 12) = *(const f32x4*)(p + 20);
  union { v16bf v; __bf16 e[16]; } u;
#pragma unroll
  for (int e = 0; e < 16; ++e) u.e[e] = (__bf16)t[e];
  return u.v;
}

// ---------------------------------------------------------------------------
// Kernel 0: one-time weight prep -> bf16, transposed so K is contiguous.
// ---------------------------------------------------------------------------
__global__ __launch_bounds__(256) void prep_weights_kernel(
    const float* __restrict__ Wq, const float* __restrict__ Wk,
    const float* __restrict__ Wv, const float* __restrict__ Wp,
    __bf16* __restrict__ WqT, __bf16* __restrict__ WkT,
    __bf16* __restrict__ WvT, __bf16* __restrict__ WpT)
{
  const int r = blockIdx.x;
  const float* src; size_t stride; __bf16* dst;
  if (r < 1024)        { int h = r >> 6, d = r & 63;
                         src = Wq + (size_t)h * C_ * D_ + d; stride = D_;
                         dst = WqT + (size_t)r * C_; }
  else if (r < 1088)   { int d = r - 1024;
                         src = Wk + d; stride = D_;
                         dst = WkT + (size_t)d * C_; }
  else if (r < 1152)   { int d = r - 1088;
                         src = Wv + d; stride = D_;
                         dst = WvT + (size_t)d * C_; }
  else                 { int n = r - 1152;
                         src = Wp + n; stride = C_;
                         dst = WpT + (size_t)n * C_; }
#pragma unroll
  for (int i = 0; i < 4; ++i) {
    int c = threadIdx.x + i * 256;
    dst[c] = (__bf16)src[(size_t)c * stride];
  }
}

// ---------------------------------------------------------------------------
// Kernel 1: fused Q/K/V projections.  grid.z in [0, H+2).
// The 64x32 weight tile is block-uniform -> double-buffered async copy into
// LDS (one b128 per thread per K-step); waves read B fragments via ds_b128.
// V is written TRANSPOSED [B][D][T] for key-contiguous PV fragments.
// ---------------------------------------------------------------------------
__global__ __launch_bounds__(256) void qkv_proj_kernel(
    const float* __restrict__ x, const __bf16* __restrict__ WqT,
    const __bf16* __restrict__ WkT, const __bf16* __restrict__ WvT,
    __bf16* __restrict__ qb, __bf16* __restrict__ kb, __bf16* __restrict__ vt)
{
  __shared__ __bf16 wtile[2][64 * 32];              // 2 x 4 KB
  const int lane = threadIdx.x & 31;
  const int wave = threadIdx.x >> 5;
  const int half = lane >> 4;
  const int l15  = lane & 15;
  const int row0 = (blockIdx.x * 8 + wave) * 16;    // row in [0, B*T)
  const int z    = blockIdx.z;

  const __bf16* WT;                                 // [64 rows][C_] K-contig
  if (z < H_)       WT = WqT + (size_t)z * D_ * C_;
  else if (z == H_) WT = WkT;
  else              WT = WvT;

  // this thread's slice of the cooperative tile copy
  const int crow = threadIdx.x >> 2;                // 0..63
  const int cch  = (threadIdx.x & 3) * 8;           // 0,8,16,24
  const __bf16* gsrc = WT + (size_t)crow * C_ + cch;

  cp16_g2l(&wtile[0][crow * 32 + cch], gsrc);       // preload kk = 0
  wait_async_lds();
  __syncthreads();

  v8f acc[4] = {};
  const float* xr = x + (size_t)(row0 + l15) * C_ + half * 8;

  int cur = 0;
  for (int kk = 0; kk < C_; kk += 32) {
    if (kk + 32 < C_)                               // prefetch next tile
      cp16_g2l(&wtile[cur ^ 1][crow * 32 + cch], gsrc + kk + 32);

    v16bf a = ldfrag_f32(xr + kk);
#pragma unroll
    for (int c = 0; c < 4; ++c) {
      v16bf bw = ldfrag_lds(&wtile[cur][(c * 16 + l15) * 32 + half * 8]);
      acc[c] = __builtin_amdgcn_wmma_f32_16x16x32_bf16(
          false, a, false, bw, (short)0, acc[c], false, false);
    }

    wait_async_lds();
    __syncthreads();
    cur ^= 1;
  }

  const int b  = row0 >> 11;                        // T_ = 2048
  const int t0 = row0 & (T_ - 1);

  if (z <= H_) {                                    // q / k: row-major
    __bf16* outp = (z < H_)
        ? qb + ((size_t)(b * H_ + z) * T_ + t0) * D_
        : kb + ((size_t)b * T_ + t0) * D_;
#pragma unroll
    for (int c = 0; c < 4; ++c)
#pragma unroll
      for (int r = 0; r < 8; ++r)
        outp[(size_t)(half * 8 + r) * D_ + c * 16 + l15] = (__bf16)acc[c][r];
  } else {                                          // v transposed [b][d][t]
    __bf16* outp = vt + (size_t)b * D_ * T_ + t0;
#pragma unroll
    for (int c = 0; c < 4; ++c)
#pragma unroll
      for (int r = 0; r < 8; ++r)
        outp[(size_t)(c * 16 + l15) * T_ + half * 8 + r] = (__bf16)acc[c][r];
  }
}

// ---------------------------------------------------------------------------
// Kernel 2: causal flash attention (MQA).  grid = (T/128, H, B).
// Per wave: 16 query rows, stream 32-key tiles, fp32 online softmax.
// ---------------------------------------------------------------------------
__global__ __launch_bounds__(256) void flash_attn_kernel(
    const __bf16* __restrict__ qb, const __bf16* __restrict__ kb,
    const __bf16* __restrict__ vt, __bf16* __restrict__ ao)
{
  __shared__ __bf16 plds[8][16 * 32];   // per-wave P transpose staging
  const int lane = threadIdx.x & 31;
  const int wave = threadIdx.x >> 5;
  const int half = lane >> 4;
  const int l15  = lane & 15;
  const int h    = blockIdx.y;
  const int b    = blockIdx.z;
  const int r0   = (blockIdx.x * 8 + wave) * 16;

  const __bf16* qp = qb + (size_t)(b * H_ + h) * T_ * D_;
  const __bf16* kp = kb + (size_t)b * T_ * D_;
  const __bf16* vp = vt + (size_t)b * D_ * T_;      // [d][t], key contig

  v16bf aq[2];
#pragma unroll
  for (int g = 0; g < 2; ++g)
    aq[g] = ldfrag(qp + (size_t)(r0 + l15) * D_ + g * 32 + half * 8);

  v8f acc[4] = {};
  float row_m[8], row_l[8];
#pragma unroll
  for (int r = 0; r < 8; ++r) { row_m[r] = -3.0e38f; row_l[r] = 0.f; }

  const float scale = 0.125f;           // D^-0.5

  for (int j0 = 0; j0 <= r0 + 15; j0 += 32) {
    __builtin_prefetch(kp + (size_t)(j0 + 32) * D_, 0, 0);

    // ---- S = Q K^T, 16x32 tile ----
    v8f S[2] = {};
#pragma unroll
    for (int ns = 0; ns < 2; ++ns)
#pragma unroll
      for (int g = 0; g < 2; ++g) {
        v16bf bk = ldfrag(kp + (size_t)(j0 + ns * 16 + l15) * D_ + g * 32 + half * 8);
        S[ns] = __builtin_amdgcn_wmma_f32_16x16x32_bf16(
            false, aq[g], false, bk, (short)0, S[ns], false, false);
      }

    // ---- scale + causal mask + tile row-max ----
    float tmax[8];
#pragma unroll
    for (int r = 0; r < 8; ++r) {
      int qrow = r0 + half * 8 + r;
#pragma unroll
      for (int ns = 0; ns < 2; ++ns) {
        int key = j0 + ns * 16 + l15;
        float s = S[ns][r] * scale;
        S[ns][r] = (key <= qrow) ? s : -3.0e38f;
      }
      float m = fmaxf(S[0][r], S[1][r]);
#pragma unroll
      for (int mk = 8; mk >= 1; mk >>= 1) m = fmaxf(m, __shfl_xor(m, mk, 32));
      tmax[r] = m;
    }

    // ---- online softmax update ----
#pragma unroll
    for (int r = 0; r < 8; ++r) {
      float mnew = fmaxf(row_m[r], tmax[r]);
      float corr = __expf(row_m[r] - mnew);
      row_m[r] = mnew;
      row_l[r] *= corr;
#pragma unroll
      for (int c = 0; c < 4; ++c) acc[c][r] *= corr;
      float p0 = __expf(S[0][r] - mnew);
      float p1 = __expf(S[1][r] - mnew);
      S[0][r] = p0; S[1][r] = p1;
      float ps = p0 + p1;
#pragma unroll
      for (int mk = 8; mk >= 1; mk >>= 1) ps += __shfl_xor(ps, mk, 32);
      row_l[r] += ps;
    }

    // ---- P: C-layout -> A-layout via per-wave LDS tile ----
    __bf16* pl = plds[wave];
#pragma unroll
    for (int r = 0; r < 8; ++r) {
      pl[(half * 8 + r) * 32 + l15]      = (__bf16)S[0][r];
      pl[(half * 8 + r) * 32 + 16 + l15] = (__bf16)S[1][r];
    }
    v16bf ap = ldfrag_lds(pl + l15 * 32 + half * 8);

    // ---- acc += P V ----
#pragma unroll
    for (int c = 0; c < 4; ++c) {
      v16bf bv = ldfrag(vp + (size_t)(c * 16 + l15) * T_ + j0 + half * 8);
      acc[c] = __builtin_amdgcn_wmma_f32_16x16x32_bf16(
          false, ap, false, bv, (short)0, acc[c], false, false);
    }
  }

  // ---- normalize + store: ao[b][t][h*64 + d] bf16 ----
  __bf16* aop = ao + (size_t)(b * T_ + r0) * C_ + h * D_;
#pragma unroll
  for (int c = 0; c < 4; ++c)
#pragma unroll
    for (int r = 0; r < 8; ++r)
      aop[(size_t)(half * 8 + r) * C_ + c * 16 + l15] =
          (__bf16)(acc[c][r] / row_l[r]);
}

// ---------------------------------------------------------------------------
// Kernel 3: output projection out = ao @ Wp + bp (fp32 out).
// Same double-buffered async LDS staging of the block-uniform WpT tile.
// ---------------------------------------------------------------------------
__global__ __launch_bounds__(256) void out_proj_kernel(
    const __bf16* __restrict__ ao, const __bf16* __restrict__ WpT,
    const float* __restrict__ bp, float* __restrict__ out)
{
  __shared__ __bf16 wtile[2][64 * 32];
  const int lane = threadIdx.x & 31;
  const int wave = threadIdx.x >> 5;
  const int half = lane >> 4;
  const int l15  = lane & 15;
  const int row0 = (blockIdx.x * 8 + wave) * 16;    // M tile
  const int n0   = blockIdx.y * 64;                 // N group

  const int crow = threadIdx.x >> 2;
  const int cch  = (threadIdx.x & 3) * 8;
  const __bf16* gsrc = WpT + (size_t)(n0 + crow) * C_ + cch;

  cp16_g2l(&wtile[0][crow * 32 + cch], gsrc);
  wait_async_lds();
  __syncthreads();

  v8f acc[4] = {};
  const __bf16* ar = ao + (size_t)(row0 + l15) * C_ + half * 8;

  int cur = 0;
  for (int kk = 0; kk < C_; kk += 32) {
    if (kk + 32 < C_)
      cp16_g2l(&wtile[cur ^ 1][crow * 32 + cch], gsrc + kk + 32);

    v16bf a = ldfrag(ar + kk);
#pragma unroll
    for (int c = 0; c < 4; ++c) {
      v16bf bw = ldfrag_lds(&wtile[cur][(c * 16 + l15) * 32 + half * 8]);
      acc[c] = __builtin_amdgcn_wmma_f32_16x16x32_bf16(
          false, a, false, bw, (short)0, acc[c], false, false);
    }

    wait_async_lds();
    __syncthreads();
    cur ^= 1;
  }

#pragma unroll
  for (int c = 0; c < 4; ++c) {
    float bias = bp[n0 + c * 16 + l15];
#pragma unroll
    for (int r = 0; r < 8; ++r)
      out[(size_t)(row0 + half * 8 + r) * C_ + n0 + c * 16 + l15] =
          acc[c][r] + bias;
  }
}

// ---------------------------------------------------------------------------
extern "C" void kernel_launch(void* const* d_in, const int* in_sizes, int n_in,
                              void* d_out, int out_size, void* d_ws, size_t ws_size,
                              hipStream_t stream) {
  const float* x  = (const float*)d_in[0];
  const float* Wq = (const float*)d_in[1];
  const float* Wk = (const float*)d_in[2];
  const float* Wv = (const float*)d_in[3];
  const float* Wp = (const float*)d_in[4];
  const float* bp = (const float*)d_in[5];
  float* out = (float*)d_out;

  // bf16 workspace (~40 MB)
  __bf16* qb  = (__bf16*)d_ws;                       // [B,H,T,D]
  __bf16* kb  = qb  + (size_t)B_ * H_ * T_ * D_;     // [B,T,D]
  __bf16* vt  = kb  + (size_t)B_ * T_ * D_;          // [B,D,T] (transposed)
  __bf16* ao  = vt  + (size_t)B_ * T_ * D_;          // [B,T,C]
  __bf16* WqT = ao  + (size_t)B_ * T_ * C_;          // [H*D, C]
  __bf16* WkT = WqT + (size_t)H_ * D_ * C_;          // [D, C]
  __bf16* WvT = WkT + (size_t)D_ * C_;               // [D, C]
  __bf16* WpT = WvT + (size_t)D_ * C_;               // [C, C]

  dim3 blk(256);

  prep_weights_kernel<<<dim3(2176), blk, 0, stream>>>(
      Wq, Wk, Wv, Wp, WqT, WkT, WvT, WpT);

  dim3 g1((B_ * T_) / (16 * 8), 1, H_ + 2);          // 64 x 1 x 18
  qkv_proj_kernel<<<g1, blk, 0, stream>>>(x, WqT, WkT, WvT, qb, kb, vt);

  dim3 g2(T_ / (16 * 8), H_, B_);                    // 16 x 16 x 4
  flash_attn_kernel<<<g2, blk, 0, stream>>>(qb, kb, vt, ao);

  dim3 g3((B_ * T_) / (16 * 8), C_ / 64);            // 64 x 16
  out_proj_kernel<<<g3, blk, 0, stream>>>(ao, WpT, bp, out);
}
